// CausalSelfAttention_33251636806063
// MI455X (gfx1250) — compile-verified, matrix operand order checked by reading, not run on hardware
//
#include <hip/hip_runtime.h>
#include <hip/hip_bf16.h>

// ---------------------------------------------------------------------------
// CDNA5 (gfx1250) causal self-attention block, bf16 WMMA everywhere.
// GEMMs: LDS-staged B tiles (async-to-LDS when available), 32x64 per wave.
// ---------------------------------------------------------------------------

#define N_TOK 8192
#define DIMC  1024
#define NH    16
#define NKV   4
#define HD    64
#define BB    8
#define SS    1024
#define QKVN  1536           // 1024 q + 256 k + 256 v output columns

typedef __attribute__((ext_vector_type(16))) __bf16 v16bf;
typedef __attribute__((ext_vector_type(8)))  __bf16 v8bf;
typedef __attribute__((ext_vector_type(4)))  __bf16 v4bf;
typedef __attribute__((ext_vector_type(8)))  float  v8f;

union Frag16 { v16bf v; v8bf h[2]; };

// ---- CDNA5 async global->LDS path (guarded; fallback = sync staging) ------
#if defined(__has_builtin)
#if __has_builtin(__builtin_amdgcn_global_load_async_to_lds_b128)
#define HAVE_ASYNC_LDS 1
#endif
#endif
#ifndef HAVE_ASYNC_LDS
#define HAVE_ASYNC_LDS 0
#endif

// Builtin prototype (from hipcc diagnostic): param0 = int4 addrspace(1)*,
// param1 = int4 addrspace(3)*, then two constant ints (offset, cpol).
typedef int vec4i __attribute__((vector_size(16)));
typedef __attribute__((address_space(1))) vec4i* as1_v4i;
typedef __attribute__((address_space(3))) vec4i* as3_v4i;

__device__ __forceinline__ void stage_chunk16(const __bf16* g, __bf16* l) {
#if HAVE_ASYNC_LDS
  __builtin_amdgcn_global_load_async_to_lds_b128(
      (as1_v4i)(vec4i*)const_cast<__bf16*>(g), (as3_v4i)(vec4i*)l, 0, 0);
#else
  *(v8bf*)l = *(const v8bf*)g;
#endif
}

#if HAVE_ASYNC_LDS
#define ASYNC_WAIT(n) asm volatile("s_wait_asynccnt %0" ::"n"(n) : "memory")
#else
#define ASYNC_WAIT(n) do { } while (0)
#endif

__device__ __forceinline__ __bf16 f2bf(float f) {
  unsigned u = __builtin_bit_cast(unsigned, f);
  unsigned r = (u + 0x7FFFu + ((u >> 16) & 1u)) >> 16;
  unsigned short h = (unsigned short)r;
  return __builtin_bit_cast(__bf16, h);
}

// Load a 16x32 bf16 A/B fragment from a row-major [rows x ld] matrix.
// Lane layout (ISA 7.12.2, 16-bit A-matrix 16x32):
//   lanes 0-15 : row = row0+lane,    K = {k0..k0+7} U {k0+16..k0+23}
//   lanes 16-31: row = row0+lane-16, K = {k0+8..k0+15} U {k0+24..k0+31}
__device__ __forceinline__ v16bf load_frag(const __bf16* M, int row0, int k0, int ld) {
  const int lane = threadIdx.x & 31;
  const int half = lane >> 4;
  const int l16  = lane & 15;
  const __bf16* p = M + (size_t)(row0 + l16) * ld + k0 + half * 8;
  Frag16 f;
  f.h[0] = *(const v8bf*)(p);
  f.h[1] = *(const v8bf*)(p + 16);
  return f.v;
}

__device__ __forceinline__ v8f wmma_bf16(v16bf a, v16bf b, v8f c) {
  return __builtin_amdgcn_wmma_f32_16x16x32_bf16(false, a, false, b,
                                                 (short)0, c, false, false);
}

// ---------------------------------------------------------------------------
// f32 -> bf16 cast, 4 elements / thread
// ---------------------------------------------------------------------------
__global__ void __launch_bounds__(256)
cvt_f32_bf16(const float* __restrict__ s, __bf16* __restrict__ d, int n4) {
  int i = blockIdx.x * 256 + threadIdx.x;
  if (i >= n4) return;
  float4 f = ((const float4*)s)[i];
  v4bf o;
  o[0] = f2bf(f.x); o[1] = f2bf(f.y); o[2] = f2bf(f.z); o[3] = f2bf(f.w);
  ((v4bf*)d)[i] = o;
}

// ---------------------------------------------------------------------------
// bf16 GEMM:  C[M x ldc](f32) = A[M x K] @ W[N x K]^T
// block = 128x64 tile, 4 waves; wave w -> rows [128*bx + 32w, +32) (2 slabs).
// B tile (64 rows x 32 k = 4 KB) double-buffered in LDS, filled with
// async global->LDS loads (ASYNCcnt) when the toolchain exposes them.
// ---------------------------------------------------------------------------
__global__ void __launch_bounds__(128)
gemm_bf16(const __bf16* __restrict__ A, const __bf16* __restrict__ W,
          float* __restrict__ C, int Kdim, int ldc) {
  __shared__ __bf16 bstage[2][64 * 32];

  const int tid  = threadIdx.x;
  const int wave = tid >> 5;
  const int lane = tid & 31;
  const int half = lane >> 4;
  const int l16  = lane & 15;
  const int m0 = blockIdx.x * 128 + wave * 32;
  const int n0 = blockIdx.y * 64;

  // Staging: 256 chunks of 16 B; thread t owns chunks t and t+128.
  // chunk c -> B-row (c>>2), in-row elem offset (c&3)*8. LDS layout [64][32].
  const __bf16* Wt = W + (size_t)n0 * Kdim;
  const int c0 = tid, c1 = tid + 128;

#define STAGE(buf, kk)                                                        \
  do {                                                                        \
    stage_chunk16(Wt + (size_t)(c0 >> 2) * Kdim + (kk) + (c0 & 3) * 8,        \
                  &bstage[buf][c0 * 8]);                                      \
    stage_chunk16(Wt + (size_t)(c1 >> 2) * Kdim + (kk) + (c1 & 3) * 8,        \
                  &bstage[buf][c1 * 8]);                                      \
  } while (0)

  v8f acc[2][4] = {};
  const int nk = Kdim / 32;

  STAGE(0, 0);
  for (int i = 0; i < nk; ++i) {
    const int k = i * 32;
    if (i + 1 < nk) {
      STAGE((i + 1) & 1, k + 32);   // prefetch next B tile (async)
      ASYNC_WAIT(2);                // current tile drained, next in flight
    } else {
      ASYNC_WAIT(0);
    }
    __syncthreads();                // tile i visible to all waves

    v16bf a0 = load_frag(A, m0, k, Kdim);
    v16bf a1 = load_frag(A, m0 + 16, k, Kdim);
    if (i + 2 < nk)                 // keep the A stream warm in L2/WGP$
      __builtin_prefetch(A + (size_t)(m0 + l16) * Kdim + k + 64, 0, 1);

    const __bf16* bt = &bstage[i & 1][0];
#pragma unroll
    for (int nt = 0; nt < 4; ++nt) {
      Frag16 bf;
      const __bf16* p = bt + (nt * 16 + l16) * 32 + half * 8;
      bf.h[0] = *(const v8bf*)(p);
      bf.h[1] = *(const v8bf*)(p + 16);
      acc[0][nt] = wmma_bf16(a0, bf.v, acc[0][nt]);
      acc[1][nt] = wmma_bf16(a1, bf.v, acc[1][nt]);
    }
    __syncthreads();                // done reading buf before its reuse
  }
#undef STAGE

#pragma unroll
  for (int s = 0; s < 2; ++s)
#pragma unroll
    for (int nt = 0; nt < 4; ++nt)
#pragma unroll
      for (int g = 0; g < 8; ++g)
        C[(size_t)(m0 + s * 16 + g + half * 8) * ldc + n0 + nt * 16 + l16] =
            acc[s][nt][g];
}

// ---------------------------------------------------------------------------
// QKV epilogue: RMSNorm + RoPE (+ q_gain), write bf16 Q/K and transposed V.
// One wave per (token, slot); slots 0-15 = q heads, 16-19 = k, 20-23 = v.
// ---------------------------------------------------------------------------
__global__ void __launch_bounds__(256)
qkv_epilogue(const float* __restrict__ qkv,
             const float* __restrict__ qgain,
             const float* __restrict__ qscale_p,
             const float* __restrict__ kscale_p,
             const float* __restrict__ rcos,
             const float* __restrict__ rsin,
             __bf16* __restrict__ qb, __bf16* __restrict__ kb,
             __bf16* __restrict__ vtb) {
  const int wid  = blockIdx.x * 8 + (threadIdx.x >> 5);
  const int lane = threadIdx.x & 31;
  const int tok  = wid / 24;
  const int slot = wid % 24;
  const int b = tok >> 10;          // / SS
  const int s = tok & (SS - 1);

  const float* row;
  if (slot < 16)      row = qkv + (size_t)tok * QKVN + slot * HD;
  else if (slot < 20) row = qkv + (size_t)tok * QKVN + 1024 + (slot - 16) * HD;
  else                row = qkv + (size_t)tok * QKVN + 1280 + (slot - 20) * HD;

  float x1 = row[lane];
  float x2 = row[lane + 32];

  if (slot >= 20) {                 // V: plain cast, store transposed [B,HKV,HD,S]
    const int hv = slot - 20;
    __bf16* vt = vtb + (size_t)(b * NKV + hv) * HD * SS;
    vt[(size_t)lane * SS + s]        = f2bf(x1);
    vt[(size_t)(lane + 32) * SS + s] = f2bf(x2);
    return;
  }

  // RMS norm over 64 elements (full-wave reduction, 2 elems / lane)
  float ss = x1 * x1 + x2 * x2;
#pragma unroll
  for (int m = 16; m; m >>= 1) ss += __shfl_xor(ss, m, 32);
  const float scl = (slot < 16) ? *qscale_p : *kscale_p;
  const float inv = rsqrtf(ss * (1.0f / 64.0f) + 1.1920929e-07f) * scl;
  x1 *= inv; x2 *= inv;

  // RoPE: pairs (i, i+32), lane = i
  const float c  = rcos[(size_t)tok * 32 + lane];
  const float sn = rsin[(size_t)tok * 32 + lane];
  float o1 =  x1 * c + x2 * sn;
  float o2 = -x1 * sn + x2 * c;

  if (slot < 16) {
    const float g = qgain[slot];
    o1 *= g; o2 *= g;
    __bf16* q = qb + ((size_t)(b * NH + slot) * SS + s) * HD;
    q[lane] = f2bf(o1); q[lane + 32] = f2bf(o2);
  } else {
    const int hk = slot - 16;
    __bf16* kk = kb + ((size_t)(b * NKV + hk) * SS + s) * HD;
    kk[lane] = f2bf(o1); kk[lane + 32] = f2bf(o2);
  }
}

// ---------------------------------------------------------------------------
// Flash attention: grid (S/64, H, B), 4 waves = one 64-row Q tile.
// Wave w owns rows [64*qt + 16w, +16); online softmax; P via LDS exchange.
// ---------------------------------------------------------------------------
__global__ void __launch_bounds__(128)
flash_attn(const __bf16* __restrict__ qb, const __bf16* __restrict__ kb,
           const __bf16* __restrict__ vtb, __bf16* __restrict__ ob) {
  __shared__ __bf16 lds_p[4][16 * 64];

  const int wave = threadIdx.x >> 5;
  const int lane = threadIdx.x & 31;
  const int half = lane >> 4;
  const int l16  = lane & 15;
  const int qt = blockIdx.x;
  const int h  = blockIdx.y;
  const int b  = blockIdx.z;
  const int hkv = h >> 2;           // NH/NKV = 4

  const __bf16* Q = qb  + (size_t)(b * NH  + h)   * SS * HD;
  const __bf16* K = kb  + (size_t)(b * NKV + hkv) * SS * HD;
  const __bf16* V = vtb + (size_t)(b * NKV + hkv) * HD * SS;

  const int qm0 = qt * 64 + wave * 16;
  const v16bf aq0 = load_frag(Q, qm0, 0, HD);
  const v16bf aq1 = load_frag(Q, qm0, 32, HD);

  v8f o[4] = {};
  float mrow[8], lrow[8];
#pragma unroll
  for (int g = 0; g < 8; ++g) { mrow[g] = -1e30f; lrow[g] = 0.0f; }
  const float scale = 0.125f;       // HD^-0.5

  for (int kt = 0; kt <= qt; ++kt) {
    const int kk = kt * 64;
    const bool diag = (kt == qt);

    // --- S = Q @ K^T (per wave: 16x64 scores) ---
    v8f st[4];
#pragma unroll
    for (int nt = 0; nt < 4; ++nt) {
      v16bf bk0 = load_frag(K, kk + nt * 16, 0, HD);
      v16bf bk1 = load_frag(K, kk + nt * 16, 32, HD);
      v8f z = {};
      z = wmma_bf16(aq0, bk0, z);
      st[nt] = wmma_bf16(aq1, bk1, z);
    }

    // --- scale + causal mask on diagonal tile ---
#pragma unroll
    for (int nt = 0; nt < 4; ++nt)
#pragma unroll
      for (int g = 0; g < 8; ++g) {
        float v = st[nt][g] * scale;
        if (diag) {
          const int qpos = qm0 + g + half * 8;
          const int kpos = kk + nt * 16 + l16;
          if (kpos > qpos) v = -1e30f;
        }
        st[nt][g] = v;
      }

    // --- online softmax: row max / exp / row sum ---
    float mnew[8], alpha[8];
#pragma unroll
    for (int g = 0; g < 8; ++g) {
      float rm = fmaxf(fmaxf(st[0][g], st[1][g]), fmaxf(st[2][g], st[3][g]));
#pragma unroll
      for (int msk = 8; msk; msk >>= 1) rm = fmaxf(rm, __shfl_xor(rm, msk, 32));
      const float mn = fmaxf(mrow[g], rm);
      mnew[g]  = mn;
      alpha[g] = __expf(mrow[g] - mn);
      mrow[g]  = mn;
    }
#pragma unroll
    for (int g = 0; g < 8; ++g) {
      float rs = 0.0f;
#pragma unroll
      for (int nt = 0; nt < 4; ++nt) {
        const float p = __expf(st[nt][g] - mnew[g]);
        rs += p;
        lds_p[wave][(g + half * 8) * 64 + nt * 16 + l16] = f2bf(p);
      }
#pragma unroll
      for (int msk = 8; msk; msk >>= 1) rs += __shfl_xor(rs, msk, 32);
      lrow[g] = lrow[g] * alpha[g] + rs;
    }

    // --- rescale running output ---
#pragma unroll
    for (int nt = 0; nt < 4; ++nt)
#pragma unroll
      for (int g = 0; g < 8; ++g) o[nt][g] *= alpha[g];

    // --- reload P as A-fragments (DS ops are in-order per wave) ---
    const __bf16* pl = &lds_p[wave][0];
    const v16bf ap0 = load_frag(pl, 0, 0, 64);
    const v16bf ap1 = load_frag(pl, 0, 32, 64);

    // --- O += P @ V  (V transposed: rows = head-dim, contiguous keys) ---
#pragma unroll
    for (int nt = 0; nt < 4; ++nt) {
      v16bf bv0 = load_frag(V, nt * 16, kk, SS);
      v16bf bv1 = load_frag(V, nt * 16, kk + 32, SS);
      o[nt] = wmma_bf16(ap0, bv0, o[nt]);
      o[nt] = wmma_bf16(ap1, bv1, o[nt]);
    }
  }

  // --- finalize: divide by row sum, store bf16 [token, H*HD] ---
#pragma unroll
  for (int nt = 0; nt < 4; ++nt)
#pragma unroll
    for (int g = 0; g < 8; ++g) {
      const float val = o[nt][g] / lrow[g];
      const int tok = b * SS + qt * 64 + wave * 16 + g + half * 8;
      const int col = h * HD + nt * 16 + l16;
      ob[(size_t)tok * DIMC + col] = f2bf(val);
    }
}

// ---------------------------------------------------------------------------
// Host launcher
// ---------------------------------------------------------------------------
extern "C" void kernel_launch(void* const* d_in, const int* in_sizes, int n_in,
                              void* d_out, int out_size, void* d_ws, size_t ws_size,
                              hipStream_t stream) {
  const float* x      = (const float*)d_in[0];
  const float* Wq     = (const float*)d_in[1];
  const float* Wk     = (const float*)d_in[2];
  const float* Wv     = (const float*)d_in[3];
  const float* Wp     = (const float*)d_in[4];
  const float* qgain  = (const float*)d_in[5];
  const float* qscale = (const float*)d_in[6];
  const float* kscale = (const float*)d_in[7];
  const float* rcos   = (const float*)d_in[8];
  const float* rsin   = (const float*)d_in[9];
  // d_in[10] cu_seqlens, d_in[11] max_doc_len: fixed-shape, unused.

  char* ws = (char*)d_ws;
  size_t off = 0;
  auto alloc = [&](size_t bytes) -> void* {
    void* p = ws + off;
    off = (off + bytes + 255) & ~(size_t)255;
    return p;
  };

  __bf16* xb   = (__bf16*)alloc((size_t)N_TOK * DIMC * 2);
  __bf16* wb   = (__bf16*)alloc((size_t)2560 * DIMC * 2);   // [Wq;Wk;Wv;Wproj]
  float*  qkv  = (float*) alloc((size_t)N_TOK * QKVN * 4);
  __bf16* qbuf = (__bf16*)alloc((size_t)BB * NH  * SS * HD * 2);
  __bf16* kbuf = (__bf16*)alloc((size_t)BB * NKV * SS * HD * 2);
  __bf16* vtb  = (__bf16*)alloc((size_t)BB * NKV * HD * SS * 2);
  __bf16* abuf = (__bf16*)alloc((size_t)N_TOK * DIMC * 2);
  (void)ws_size; (void)in_sizes; (void)n_in; (void)out_size;

  // 1) casts
  cvt_f32_bf16<<<(N_TOK * DIMC / 4 + 255) / 256, 256, 0, stream>>>(x, xb, N_TOK * DIMC / 4);
  cvt_f32_bf16<<<(1024 * 1024 / 4 + 255) / 256, 256, 0, stream>>>(Wq, wb, 1024 * 1024 / 4);
  cvt_f32_bf16<<<(256 * 1024 / 4 + 255) / 256, 256, 0, stream>>>(Wk, wb + (size_t)1024 * 1024, 256 * 1024 / 4);
  cvt_f32_bf16<<<(256 * 1024 / 4 + 255) / 256, 256, 0, stream>>>(Wv, wb + (size_t)1280 * 1024, 256 * 1024 / 4);
  cvt_f32_bf16<<<(1024 * 1024 / 4 + 255) / 256, 256, 0, stream>>>(Wp, wb + (size_t)1536 * 1024, 1024 * 1024 / 4);

  // 2) fused QKV projection: [N,1536] = xb @ [Wq;Wk;Wv]^T
  gemm_bf16<<<dim3(N_TOK / 128, QKVN / 64), 128, 0, stream>>>(xb, wb, qkv, DIMC, QKVN);

  // 3) RMSNorm + RoPE + gain; layout Q/K/V for attention
  qkv_epilogue<<<(N_TOK * 24) / 8, 256, 0, stream>>>(qkv, qgain, qscale, kscale,
                                                     rcos, rsin, qbuf, kbuf, vtb);

  // 4) causal flash attention
  flash_attn<<<dim3(SS / 64, NH, BB), 128, 0, stream>>>(qbuf, kbuf, vtb, abuf);

  // 5) output projection -> f32 d_out
  gemm_bf16<<<dim3(N_TOK / 128, DIMC / 64), 128, 0, stream>>>(abuf, wb + (size_t)1536 * 1024,
                                                              (float*)d_out, DIMC, DIMC);
}
